// InterestDictSoft_uni_71511205478462
// MI455X (gfx1250) — matrix operand order repaired and checked by reading, not compiled, not run on hardware
//
#include <hip/hip_runtime.h>
#include <hip/hip_bf16.h>

// ---------------------------------------------------------------------------
// Soft VQ dictionary kernel for MI455X (gfx1250, wave32, WMMA).
// Fused plan: bf16 WMMA GEMM (X @ DictT) -> cosine -> distances write +
// on-chip top-5 -> softmax / group_emb / scatter, then masked EMA update.
// Roofline: ~330 MB of unavoidable HBM traffic (256 MB = distances output)
// dominates; bf16 WMMA keeps the 17.2 GFLOP GEMM under the 23.3 TB/s roof.
// ---------------------------------------------------------------------------

typedef __bf16 bf16_t;
typedef __bf16 v16bf __attribute__((ext_vector_type(16)));
typedef __bf16 v8bfv __attribute__((ext_vector_type(8)));
typedef __bf16 v4bfv __attribute__((ext_vector_type(4)));
typedef float  v8f   __attribute__((ext_vector_type(8)));

#define N_  65536
#define M_  1024
#define D_  128
#define K_  5
#define DECAY_ 0.1f
#define TEMP_  20.0f
#define EPS_   1e-8f

// ---------------------------------------------------------------------------
// Zero the scatter workspace (cluster_size + input_sum).
// ---------------------------------------------------------------------------
__global__ void k_zero(float* __restrict__ p, int n) {
  int i = blockIdx.x * blockDim.x + threadIdx.x;
  if (i < n) p[i] = 0.0f;
}

// ---------------------------------------------------------------------------
// fp32 -> bf16 conversion + row L2 norms. One wave32 per row (D=128 -> 4/lane).
// ---------------------------------------------------------------------------
__global__ void k_prep(const float* __restrict__ src, bf16_t* __restrict__ dst,
                       float* __restrict__ norms, int nrows) {
  const int wave = threadIdx.x >> 5;
  const int lane = threadIdx.x & 31;
  const int row  = blockIdx.x * (blockDim.x >> 5) + wave;
  if (row >= nrows) return;
  const float4 x = ((const float4*)(src + (size_t)row * D_))[lane];
  v4bfv o;
  o[0] = (bf16_t)x.x; o[1] = (bf16_t)x.y; o[2] = (bf16_t)x.z; o[3] = (bf16_t)x.w;
  *(v4bfv*)(dst + (size_t)row * D_ + lane * 4) = o;
  float ss = x.x * x.x + x.y * x.y + x.z * x.z + x.w * x.w;
  #pragma unroll
  for (int off = 16; off > 0; off >>= 1) ss += __shfl_down(ss, off);
  if (lane == 0) norms[row] = sqrtf(ss);
}

// ---------------------------------------------------------------------------
// Main fused kernel. 256 threads = 8 waves; each wave owns a 16-row stripe
// and sweeps all 64 column tiles (16 dict rows each) with 4 bf16 WMMAs/tile.
// ---------------------------------------------------------------------------
__global__ __launch_bounds__(256) void k_main(
    const bf16_t* __restrict__ xb,   const bf16_t* __restrict__ db,
    const float*  __restrict__ xnorm, const float* __restrict__ dnorm,
    const float*  __restrict__ X,     const float* __restrict__ Dict,
    float* __restrict__ gout, int* __restrict__ tki, float* __restrict__ dist,
    float* __restrict__ csize, float* __restrict__ isum) {

  __shared__ float tileS[8][16][17];   // padded: conflict-free row scans
  __shared__ float wS[8][16][K_];
  __shared__ float eS[8][16][K_];
  __shared__ int   iS[8][16][K_];

  const int wave = threadIdx.x >> 5;
  const int lane = threadIdx.x & 31;
  const int g    = lane >> 4;          // lane group (0: lanes 0-15, 1: 16-31)
  const int r    = lane & 15;
  const int row0 = (blockIdx.x * 8 + wave) * 16;

  // ---- A fragments: 16 rows x 128 K of bf16 X (two 16B chunks per K-step) --
  union Frag { v16bf v; v8bfv h[2]; };
  Frag a[4];
  {
    const v8bfv* xr = (const v8bfv*)(xb + (size_t)(row0 + r) * D_);
    #pragma unroll
    for (int kk = 0; kk < 4; ++kk) {        // A 16x32: K = kk*32 + g*8 and +16
      a[kk].h[0] = xr[kk * 4 + g];
      a[kk].h[1] = xr[kk * 4 + g + 2];
    }
  }
  float xn[8];
  #pragma unroll
  for (int v = 0; v < 8; ++v) xn[v] = xnorm[row0 + 8 * g + v];   // C-layout rows

  float tv[K_]; int ti[K_];
  #pragma unroll
  for (int k = 0; k < K_; ++k) { tv[k] = -2.0f; ti[k] = 0; }

  for (int ct = 0; ct < M_ / 16; ++ct) {
    const int n0 = ct * 16;
    const bf16_t* drow = db + (size_t)(n0 + r) * D_;
    if (ct + 1 < M_ / 16) __builtin_prefetch(drow + 16 * D_, 0, 3);

    // ---- B fragments: row-major Dict rows == DictT columns ----------------
    Frag b[4];
    const v8bfv* br = (const v8bfv*)drow;
    #pragma unroll
    for (int kk = 0; kk < 4; ++kk) {        // B 32x16: K = kk*32 + g*8 and +16
      b[kk].h[0] = br[kk * 4 + g];
      b[kk].h[1] = br[kk * 4 + g + 2];
    }

    v8f acc = {};
    #pragma unroll
    for (int kk = 0; kk < 4; ++kk)
      acc = __builtin_amdgcn_wmma_f32_16x16x32_bf16(
          false, a[kk].v, false, b[kk].v, (short)0, acc, false, false);

    const float dn = dnorm[n0 + r];
    #pragma unroll
    for (int v = 0; v < 8; ++v) {
      // native v_rcp_f32 (1 ulp) instead of IEEE divide: cosine sims don't
      // need correct rounding, and this keeps the WMMA->store path short.
      float inv = __builtin_amdgcn_rcpf(fmaxf(xn[v] * dn, EPS_));
      float d   = acc[v] * inv;
      dist[(size_t)(row0 + 8 * g + v) * M_ + n0 + r] = d;
      tileS[wave][8 * g + v][r] = d;
    }
    asm volatile("s_wait_dscnt 0" ::: "memory");

    if (g == 0) {                      // lane r maintains top-5 for its row
      #pragma unroll
      for (int c = 0; c < 16; ++c) {
        float v = tileS[wave][r][c];
        int   ix = n0 + c;
        if (v > tv[4]) {
          if (v > tv[3]) { tv[4] = tv[3]; ti[4] = ti[3];
            if (v > tv[2]) { tv[3] = tv[2]; ti[3] = ti[2];
              if (v > tv[1]) { tv[2] = tv[1]; ti[2] = ti[1];
                if (v > tv[0]) { tv[1] = tv[0]; ti[1] = ti[0]; tv[0] = v; ti[0] = ix; }
                else           { tv[1] = v; ti[1] = ix; }
              } else { tv[2] = v; ti[2] = ix; }
            } else { tv[3] = v; ti[3] = ix; }
          } else { tv[4] = v; ti[4] = ix; }
        }
      }
    }
  }

  // ---- per-row weights, topk_idx output, cluster mass ----------------------
  if (g == 0) {
    const float m0 = tv[0];
    float e[K_], se = 0.0f;
    #pragma unroll
    for (int k = 0; k < K_; ++k) { e[k] = __expf(tv[k] - m0); se += e[k]; }
    const float inv = 1.0f / se;
    #pragma unroll
    for (int k = 0; k < K_; ++k) {
      // enc_w*scale simplifies: d0 * exp(TEMP*(dk - d0))
      const float ew = m0 * __expf(TEMP_ * (tv[k] - m0));
      wS[wave][r][k] = e[k] * inv;
      eS[wave][r][k] = ew;
      iS[wave][r][k] = ti[k];
      tki[(size_t)(row0 + r) * K_ + k] = ti[k];
      atomicAdd(&csize[ti[k]], ew);
    }
  }
  asm volatile("s_wait_dscnt 0" ::: "memory");

  // ---- group_emb gather + input_sum scatter (whole wave, 4 dims/lane) ------
  for (int rr = 0; rr < 16; ++rr) {
    const int row = row0 + rr;
    const float4 xv = ((const float4*)(X + (size_t)row * D_))[lane];
    float4 ga = make_float4(0.f, 0.f, 0.f, 0.f);
    #pragma unroll
    for (int k = 0; k < K_; ++k) {
      const int   m  = iS[wave][rr][k];
      const float w  = wS[wave][rr][k];
      const float ew = eS[wave][rr][k];
      const float4 dv = ((const float4*)(Dict + (size_t)m * D_))[lane];
      ga.x += w * dv.x; ga.y += w * dv.y; ga.z += w * dv.z; ga.w += w * dv.w;
      float* ip = isum + (size_t)m * D_ + lane * 4;
      atomicAdd(ip + 0, ew * xv.x);
      atomicAdd(ip + 1, ew * xv.y);
      atomicAdd(ip + 2, ew * xv.z);
      atomicAdd(ip + 3, ew * xv.w);
    }
    ((float4*)(gout + (size_t)row * D_))[lane] = ga;
  }
}

// ---------------------------------------------------------------------------
// Masked EMA dictionary update (tiny: keep the precise divide here).
// ---------------------------------------------------------------------------
__global__ void k_ema(const float* __restrict__ dict, const float* __restrict__ csize,
                      const float* __restrict__ isum, float* __restrict__ ndict) {
  int i = blockIdx.x * blockDim.x + threadIdx.x;   // i < M_*D_
  int m = i >> 7;
  float cs = csize[m];
  float dv = dict[i];
  float out = dv;
  if (cs != 0.0f) out = dv * DECAY_ + (isum[i] / cs) * (1.0f - DECAY_);
  ndict[i] = out;
}

// ---------------------------------------------------------------------------
extern "C" void kernel_launch(void* const* d_in, const int* in_sizes, int n_in,
                              void* d_out, int out_size, void* d_ws, size_t ws_size,
                              hipStream_t stream) {
  (void)in_sizes; (void)n_in; (void)out_size; (void)ws_size;
  const float* X    = (const float*)d_in[0];   // [N, D]
  const float* Dict = (const float*)d_in[1];   // [M, D]

  // Output layout (flat, return order): group_emb, topk_idx, distances, new_dict
  float* out   = (float*)d_out;
  float* gout  = out;
  int*   tki   = (int*)(out + (size_t)N_ * D_);
  float* dist  = out + (size_t)N_ * D_ + (size_t)N_ * K_;
  float* ndict = dist + (size_t)N_ * M_;

  // Workspace layout
  char*   ws    = (char*)d_ws;
  bf16_t* xb    = (bf16_t*)ws;                                   // N*D bf16
  bf16_t* db    = (bf16_t*)(ws + (size_t)N_ * D_ * 2);           // M*D bf16
  float*  xnorm = (float*)(ws + (size_t)N_ * D_ * 2 + (size_t)M_ * D_ * 2);
  float*  dnorm = xnorm + N_;
  float*  csize = dnorm + M_;
  float*  isum  = csize + M_;                                    // M*D floats

  const int nz = M_ + M_ * D_;
  k_zero<<<(nz + 255) / 256, 256, 0, stream>>>(csize, nz);
  k_prep<<<N_ / 8, 256, 0, stream>>>(X, xb, xnorm, N_);
  k_prep<<<M_ / 8, 256, 0, stream>>>(Dict, db, dnorm, M_);
  k_main<<<N_ / (16 * 8), 256, 0, stream>>>(xb, db, xnorm, dnorm, X, Dict,
                                            gout, tki, dist, csize, isum);
  k_ema<<<(M_ * D_) / 256, 256, 0, stream>>>(Dict, csize, isum, ndict);
}
